// BiMamba_Layer_16509854286586
// MI455X (gfx1250) — matrix operand order repaired and checked by reading, not compiled, not run on hardware
//
#include <hip/hip_runtime.h>
#include <hip/hip_bf16.h>

// ---------------------------------------------------------------------------
// BiMamba layer for MI455X (gfx1250, wave32, WMMA).
// GEMMs: bf16 operands (pre-converted), v_wmma_f32_16x16x32_bf16, f32 accum,
//        b128 global->reg->LDS staging, double-buffered LDS (1 barrier/iter),
//        register-prefetch software pipeline.
// Scan/LN/activations: f32 VALU. Backward direction via index flipping.
// ---------------------------------------------------------------------------

#define DM    512
#define DI    1024
#define NST   32
#define BSZ   8
#define LSEQ  1024
#define ROWS  (BSZ * LSEQ)   // 8192
#define LN_EPS 1e-12f

typedef __attribute__((ext_vector_type(16))) __bf16 v16bf;
typedef __attribute__((ext_vector_type(8)))  __bf16 v8bf;
typedef __attribute__((ext_vector_type(8)))  float  v8f;

__device__ __forceinline__ v8bf pack8(float4 a, float4 b)
{
    v8bf o;
    o[0] = (__bf16)a.x; o[1] = (__bf16)a.y; o[2] = (__bf16)a.z; o[3] = (__bf16)a.w;
    o[4] = (__bf16)b.x; o[5] = (__bf16)b.y; o[6] = (__bf16)b.z; o[7] = (__bf16)b.w;
    return o;
}

// ===========================================================================
// f32 -> bf16 bulk convert (n must be a multiple of 8; all our sizes are)
// ===========================================================================
__global__ __launch_bounds__(256)
void cvt_bf16_kernel(const float* __restrict__ src, __bf16* __restrict__ dst, int n)
{
    const int i = (blockIdx.x * 256 + threadIdx.x) * 8;
    if (i >= n) return;
    float4 a = *reinterpret_cast<const float4*>(src + i);
    float4 b = *reinterpret_cast<const float4*>(src + i + 4);
    *reinterpret_cast<v8bf*>(dst + i) = pack8(a, b);
}

// dt slice: xdbl (ROWS,96) cols 0..31 -> dt_bf (ROWS,32)
__global__ __launch_bounds__(256)
void cvt_dt_kernel(const float* __restrict__ xdbl, __bf16* __restrict__ dt_bf)
{
    const int i = blockIdx.x * 256 + threadIdx.x;   // ROWS*4 threads
    const int r = i >> 2;
    const int c = (i & 3) * 8;
    float4 a = *reinterpret_cast<const float4*>(xdbl + r * 96 + c);
    float4 b = *reinterpret_cast<const float4*>(xdbl + r * 96 + c + 4);
    *reinterpret_cast<v8bf*>(dt_bf + r * 32 + c) = pack8(a, b);
}

// ===========================================================================
// WMMA GEMM:  C[M,N] = A[M,K]bf16 (stride lda) * W[N,K]bf16^T  (+ bias[N])
// Block tile 64x128, 8 waves, each wave 32x32 via 2x2 WMMA 16x16x32 bf16.
// Double-buffered LDS: one barrier per K tile. Register prefetch pipeline.
// Requires M % 64 == 0 (true: M = 8192) and K % 32 == 0 (true: 32..2048).
// Out-of-range W rows are clamped (not zeroed): they only feed output
// columns >= N, which are discarded by the guarded store.
// ===========================================================================
#define BM 64
#define BN 128
#define BK 32

__global__ __launch_bounds__(256)
void gemm_bf16_wmma(const __bf16* __restrict__ A, const __bf16* __restrict__ W,
                    const float* __restrict__ bias, float* __restrict__ C,
                    int M, int N, int K, int lda, int ldc)
{
    __shared__ __align__(16) __bf16 As[2][BM][BK];   // 8 KB
    __shared__ __align__(16) __bf16 Bs[2][BN][BK];   // 16 KB

    const int tid  = threadIdx.x;
    const int lane = tid & 31;
    const int wid  = tid >> 5;     // 8 waves
    const int wm   = wid >> 2;     // 0..1
    const int wn   = wid & 3;      // 0..3
    const int m0   = blockIdx.y * BM;
    const int n0   = blockIdx.x * BN;

    // staging roles: A: 8 bf16 (16B) per thread; W: 16 bf16 (32B) per thread
    const int rA = tid >> 2, cA = (tid & 3) * 8;
    const int rB = tid >> 1, cB = (tid & 1) * 16;
    const int rBc = ((n0 + rB) < N) ? (n0 + rB) : (N - 1);   // clamp, no mask
    const __bf16* pA = A + (long)(m0 + rA) * lda + cA;
    const __bf16* pB = W + (long)rBc * K + cB;

    // prefetch first K tile into registers, commit to buffer 0
    v8bf ra  = *reinterpret_cast<const v8bf*>(pA);
    v8bf rb0 = *reinterpret_cast<const v8bf*>(pB);
    v8bf rb1 = *reinterpret_cast<const v8bf*>(pB + 8);
    *reinterpret_cast<v8bf*>(&As[0][rA][cA])     = ra;
    *reinterpret_cast<v8bf*>(&Bs[0][rB][cB])     = rb0;
    *reinterpret_cast<v8bf*>(&Bs[0][rB][cB + 8]) = rb1;
    __syncthreads();

    v8f acc[2][2] = {};

    const int sub   = lane & 15;
    const int abase = (lane >= 16) ? 8  : 0;  // A frag: K {0-7,16-23}/{8-15,24-31}
    const int bbase = (lane >= 16) ? 8  : 0;  // B frag: K {0-7,16-23}/{8-15,24-31}

    const int nT = K / BK;
    for (int kt = 0; kt < nT; ++kt) {
        const int buf = kt & 1;
        const bool more = (kt + 1) < nT;

        // prefetch next K tile into registers (overlaps WMMA stage)
        if (more) {
            const int ko = (kt + 1) * BK;
            ra  = *reinterpret_cast<const v8bf*>(pA + ko);
            rb0 = *reinterpret_cast<const v8bf*>(pB + ko);
            rb1 = *reinterpret_cast<const v8bf*>(pB + ko + 8);
        }

        // fragments per ISA 7.12.2 layouts, from current buffer
        v16bf afrag[2], bfrag[2];
#pragma unroll
        for (int i = 0; i < 2; ++i) {
            const int m = wm * 32 + i * 16 + sub;        // lane = M row
            v8bf lo = *reinterpret_cast<const v8bf*>(&As[buf][m][abase]);
            v8bf hi = *reinterpret_cast<const v8bf*>(&As[buf][m][abase + 16]);
            afrag[i] = __builtin_shufflevector(lo, hi,
                0,1,2,3,4,5,6,7,8,9,10,11,12,13,14,15);
        }
#pragma unroll
        for (int j = 0; j < 2; ++j) {
            const int n = wn * 32 + j * 16 + sub;        // lane = N col
            v8bf lo = *reinterpret_cast<const v8bf*>(&Bs[buf][n][bbase]);
            v8bf hi = *reinterpret_cast<const v8bf*>(&Bs[buf][n][bbase + 16]);
            bfrag[j] = __builtin_shufflevector(lo, hi,
                0,1,2,3,4,5,6,7,8,9,10,11,12,13,14,15);
        }

#pragma unroll
        for (int i = 0; i < 2; ++i)
#pragma unroll
            for (int j = 0; j < 2; ++j)
                acc[i][j] = __builtin_amdgcn_wmma_f32_16x16x32_bf16(
                    false, afrag[i], false, bfrag[j],
                    (short)0, acc[i][j], false, false);

        // commit next tile to the other buffer (safe: everyone finished
        // reading it before the previous barrier), then publish it
        if (more) {
            *reinterpret_cast<v8bf*>(&As[buf ^ 1][rA][cA])     = ra;
            *reinterpret_cast<v8bf*>(&Bs[buf ^ 1][rB][cB])     = rb0;
            *reinterpret_cast<v8bf*>(&Bs[buf ^ 1][rB][cB + 8]) = rb1;
            __syncthreads();
        }
    }

    // store C (+bias), guard N edge
#pragma unroll
    for (int i = 0; i < 2; ++i) {
#pragma unroll
        for (int j = 0; j < 2; ++j) {
            const int col = n0 + wn * 32 + j * 16 + (lane & 15);
            const int rb  = m0 + wm * 32 + i * 16 + ((lane >= 16) ? 8 : 0);
            if (col >= N) continue;
            const float badd = bias ? bias[col] : 0.0f;
#pragma unroll
            for (int v = 0; v < 8; ++v)
                C[(long)(rb + v) * ldc + col] = acc[i][j][v] + badd;
        }
    }
}

// ===========================================================================
// Depthwise causal (dir=0) / anti-causal (dir=1) conv(4) + bias + SiLU
// writes f32 (for the scan) and bf16 (for the x_proj GEMM)
// ===========================================================================
__global__ __launch_bounds__(256)
void conv_silu_kernel(const float* __restrict__ xz, const float* __restrict__ w,
                      const float* __restrict__ bias, float* __restrict__ xc,
                      __bf16* __restrict__ xc_bf, int dir)
{
    const int i = blockIdx.x * 256 + threadIdx.x;   // over ROWS*DI
    const int d = i & (DI - 1);
    const int l = (i >> 10) & (LSEQ - 1);
    const int b = i >> 20;
    float acc = bias[d];
#pragma unroll
    for (int j = 0; j < 4; ++j) {
        const int lp = dir ? (l + 3 - j) : (l - 3 + j);
        if (lp >= 0 && lp < LSEQ)
            acc += w[d * 4 + j] * xz[((b * LSEQ + lp) * 2 * DI) + d];
    }
    const float s = acc / (1.0f + __expf(-acc));
    xc[i] = s;
    xc_bf[i] = (__bf16)s;
}

// softplus in place
__global__ __launch_bounds__(256)
void softplus_kernel(float* __restrict__ p, int n)
{
    const int i = blockIdx.x * 256 + threadIdx.x;
    if (i >= n) return;
    const float x = p[i];
    p[i] = (x > 20.0f) ? x : log1pf(__expf(x));
}

// y *= silu(z);  also emit bf16 copy for the out-projection GEMM
__global__ __launch_bounds__(256)
void mul_silu_kernel(float* __restrict__ y, const float* __restrict__ xz,
                     __bf16* __restrict__ y_bf)
{
    const int i = blockIdx.x * 256 + threadIdx.x;   // over ROWS*DI
    const int row = i >> 10;
    const int d   = i & (DI - 1);
    const float z = xz[row * (2 * DI) + DI + d];
    const float v = y[i] * (z / (1.0f + __expf(-z)));
    y[i] = v;
    y_bf[i] = (__bf16)v;
}

// exact GELU: read f32, write bf16 (only the GEMM consumes it)
__global__ __launch_bounds__(256)
void gelu_kernel(const float* __restrict__ in, __bf16* __restrict__ out, int n)
{
    const int i = blockIdx.x * 256 + threadIdx.x;
    if (i >= n) return;
    const float x = in[i];
    out[i] = (__bf16)(0.5f * x * (1.0f + erff(x * 0.70710678118654752440f)));
}

// ===========================================================================
// Selective scan: one wave32 per (b,d); lane = state index n (N=32).
// ===========================================================================
__global__ __launch_bounds__(256)
void selective_scan_kernel(const float* __restrict__ delta,
                           const float* __restrict__ xdbl,  // (ROWS,96): dt|B|C
                           const float* __restrict__ u,     // xc (f32)
                           const float* __restrict__ A_log,
                           const float* __restrict__ Dv,
                           float* __restrict__ y, int dir)
{
    const int gid = blockIdx.x * 256 + threadIdx.x;   // BSZ*DI*32 threads
    const int n = gid & 31;
    const int d = (gid >> 5) & (DI - 1);
    const int b = gid >> 15;

    const float a  = -__expf(A_log[d * NST + n]);
    const float Dd = Dv[d];
    float h = 0.0f;

    for (int tt = 0; tt < LSEQ; ++tt) {
        const int t    = dir ? (LSEQ - 1 - tt) : tt;
        const int base = b * LSEQ + t;
        const float dt = delta[base * DI + d];       // wave-broadcast
        const float ut = u[base * DI + d];           // wave-broadcast
        const float Bn = xdbl[base * 96 + 32 + n];   // coalesced
        const float Cn = xdbl[base * 96 + 64 + n];

        h = __expf(dt * a) * h + (dt * ut) * Bn;
        float p = h * Cn;
#pragma unroll
        for (int off = 16; off >= 1; off >>= 1)
            p += __shfl_xor(p, off, 32);
        if (n == 0) y[base * DI + d] = p + ut * Dd;
    }
}

// ===========================================================================
// LayerNorm kernels, one 256-thread block per row of 512
// ===========================================================================
__device__ __forceinline__ float block_sum(float v, float* red)
{
    const int tid = threadIdx.x;
    red[tid] = v;
    __syncthreads();
    for (int s = 128; s > 0; s >>= 1) {
        if (tid < s) red[tid] += red[tid + s];
        __syncthreads();
    }
    const float r = red[0];
    __syncthreads();
    return r;
}

// out1 = LN( hf*w0 + hb*w1 + pb + x )  ;  also emit bf16 copy for FFN GEMM
__global__ __launch_bounds__(256)
void combine_ln_kernel(const float* __restrict__ hf, const float* __restrict__ hb,
                       const float* __restrict__ x,  const float* __restrict__ pw,
                       const float* __restrict__ pb, const float* __restrict__ g,
                       const float* __restrict__ bt, float* __restrict__ out,
                       __bf16* __restrict__ out_bf)
{
    __shared__ float red[256];
    const int row = blockIdx.x, tid = threadIdx.x;
    const long off = (long)row * DM;
    const float w0 = pw[0], w1 = pw[1], b0 = pb[0];
    const float v0 = hf[off + tid]       * w0 + hb[off + tid]       * w1 + b0 + x[off + tid];
    const float v1 = hf[off + tid + 256] * w0 + hb[off + tid + 256] * w1 + b0 + x[off + tid + 256];
    const float mu = block_sum(v0 + v1, red) * (1.0f / DM);
    const float d0 = v0 - mu, d1 = v1 - mu;
    const float var = block_sum(d0 * d0 + d1 * d1, red) * (1.0f / DM);
    const float rs = rsqrtf(var + LN_EPS);
    const float o0 = d0 * rs * g[tid]       + bt[tid];
    const float o1 = d1 * rs * g[tid + 256] + bt[tid + 256];
    out[off + tid] = o0;            out[off + tid + 256] = o1;
    out_bf[off + tid] = (__bf16)o0; out_bf[off + tid + 256] = (__bf16)o1;
}

// out = LN( a + b )
__global__ __launch_bounds__(256)
void add_ln_kernel(const float* __restrict__ a, const float* __restrict__ b,
                   const float* __restrict__ g, const float* __restrict__ bt,
                   float* __restrict__ out)
{
    __shared__ float red[256];
    const int row = blockIdx.x, tid = threadIdx.x;
    const long off = (long)row * DM;
    const float v0 = a[off + tid]       + b[off + tid];
    const float v1 = a[off + tid + 256] + b[off + tid + 256];
    const float mu = block_sum(v0 + v1, red) * (1.0f / DM);
    const float d0 = v0 - mu, d1 = v1 - mu;
    const float var = block_sum(d0 * d0 + d1 * d1, red) * (1.0f / DM);
    const float rs = rsqrtf(var + LN_EPS);
    out[off + tid]       = d0 * rs * g[tid]       + bt[tid];
    out[off + tid + 256] = d1 * rs * g[tid + 256] + bt[tid + 256];
}

// ===========================================================================
static inline void gemm(const __bf16* A, const __bf16* W, const float* bias,
                        float* C, int M, int N, int K, int lda, int ldc,
                        hipStream_t s)
{
    dim3 grid((N + BN - 1) / BN, (M + BM - 1) / BM);
    gemm_bf16_wmma<<<grid, 256, 0, s>>>(A, W, bias, C, M, N, K, lda, ldc);
}

static inline void cvt(const float* src, __bf16* dst, int n, hipStream_t s)
{
    cvt_bf16_kernel<<<(n / 8 + 255) / 256, 256, 0, s>>>(src, dst, n);
}

extern "C" void kernel_launch(void* const* d_in, const int* in_sizes, int n_in,
                              void* d_out, int out_size, void* d_ws, size_t ws_size,
                              hipStream_t stream)
{
    const float* x = (const float*)d_in[0];
    // d_in[1] = lengths (unused by the reference body)
    const float* fp[9];
    const float* bp[9];
    for (int i = 0; i < 9; ++i) {
        fp[i] = (const float*)d_in[2 + i];
        bp[i] = (const float*)d_in[11 + i];
    }
    const float* proj_w   = (const float*)d_in[20];
    const float* proj_b   = (const float*)d_in[21];
    const float* ln_g     = (const float*)d_in[22];
    const float* ln_b     = (const float*)d_in[23];
    const float* ffn_w1   = (const float*)d_in[24];
    const float* ffn_b1   = (const float*)d_in[25];
    const float* ffn_w2   = (const float*)d_in[26];
    const float* ffn_b2   = (const float*)d_in[27];
    const float* ffn_ln_g = (const float*)d_in[28];
    const float* ffn_ln_b = (const float*)d_in[29];

    // ---- f32 workspace ----
    float* ws    = (float*)d_ws;
    float* xz    = ws;                            // ROWS*2*DI
    float* xc    = xz    + (long)ROWS * 2 * DI;   // ROWS*DI
    float* xdbl  = xc    + (long)ROWS * DI;       // ROWS*96
    float* delta = xdbl  + (long)ROWS * 96;       // ROWS*DI
    float* yb    = delta + (long)ROWS * DI;       // ROWS*DI
    float* hf    = yb    + (long)ROWS * DI;       // ROWS*DM
    float* hb    = hf    + (long)ROWS * DM;       // ROWS*DM
    float* out1  = hb    + (long)ROWS * DM;       // ROWS*DM
    float* ffnh  = xz;   // reuse
    float* ffno  = yb;   // reuse

    // ---- bf16 workspace (after f32 region) ----
    __bf16* bws     = (__bf16*)(out1 + (long)ROWS * DM);
    __bf16* x_bf    = bws;                               // ROWS*DM
    __bf16* wgt_bf  = x_bf   + (long)ROWS * DM;          // arena 2,228,224
    __bf16* xc_bf   = wgt_bf + 2228224;                  // ROWS*DI
    __bf16* dt_bf   = xc_bf  + (long)ROWS * DI;          // ROWS*32
    __bf16* out1_bf = dt_bf  + (long)ROWS * 32;          // ROWS*DM
    __bf16* ffnh_bf = out1_bf+ (long)ROWS * DM;          // ROWS*4*DM
    __bf16* y_bf    = xc_bf;  // reuse (xc_bf consumed before gate)

    // weight arena layout
    __bf16* wA = wgt_bf;              // in_w   2048*512 = 1,048,576
    __bf16* wX = wA + 1048576;        // xproj  96*1024  =    98,304
    __bf16* wD = wX + 98304;          // dt_w   1024*32  =    32,768
    __bf16* wO = wD + 32768;          // out_w  512*1024 =   524,288
    __bf16* w1 = wgt_bf;              // ffn_w1 (reuses arena after dirs)
    __bf16* w2 = wgt_bf + 1048576;    // ffn_w2

    const int nElemDI = ROWS * DI;    // 8,388,608

    // input activations once
    cvt(x, x_bf, ROWS * DM, stream);

    for (int dir = 0; dir < 2; ++dir) {
        const float* const* P = dir ? bp : fp;
        float* hout = dir ? hb : hf;

        // convert this direction's GEMM weights
        cvt(P[0], wA, 2 * DI * DM, stream);
        cvt(P[3], wX, 96 * DI, stream);
        cvt(P[4], wD, DI * NST, stream);
        cvt(P[8], wO, DM * DI, stream);

        // in_proj: (ROWS,512) x (2048,512)^T -> xz
        gemm(x_bf, wA, nullptr, xz, ROWS, 2 * DI, DM, DM, 2 * DI, stream);
        // depthwise conv + SiLU -> xc (f32) + xc_bf
        conv_silu_kernel<<<nElemDI / 256, 256, 0, stream>>>(xz, P[1], P[2],
                                                            xc, xc_bf, dir);
        // x_proj: (ROWS,1024) x (96,1024)^T -> xdbl
        gemm(xc_bf, wX, nullptr, xdbl, ROWS, 96, DI, DI, 96, stream);
        // dt slice -> bf16, then dt_proj (+ dt_b) -> delta
        cvt_dt_kernel<<<(ROWS * 4) / 256, 256, 0, stream>>>(xdbl, dt_bf);
        gemm(dt_bf, wD, P[5], delta, ROWS, DI, NST, NST, DI, stream);
        softplus_kernel<<<nElemDI / 256, 256, 0, stream>>>(delta, nElemDI);
        // selective scan
        selective_scan_kernel<<<(BSZ * DI * 32) / 256, 256, 0, stream>>>(
            delta, xdbl, xc, P[6], P[7], yb, dir);
        // gate: y *= silu(z)  (emits y_bf)
        mul_silu_kernel<<<nElemDI / 256, 256, 0, stream>>>(yb, xz, y_bf);
        // out_proj -> hf/hb
        gemm(y_bf, wO, nullptr, hout, ROWS, DM, DI, DI, DM, stream);
    }

    // combine + residual + LN -> out1 (+bf16)
    combine_ln_kernel<<<ROWS, 256, 0, stream>>>(hf, hb, x, proj_w, proj_b,
                                                ln_g, ln_b, out1, out1_bf);
    // FFN
    cvt(ffn_w1, w1, 4 * DM * DM, stream);
    cvt(ffn_w2, w2, DM * 4 * DM, stream);
    gemm(out1_bf, w1, ffn_b1, ffnh, ROWS, 4 * DM, DM, DM, 4 * DM, stream);
    gelu_kernel<<<(ROWS * 4 * DM) / 256, 256, 0, stream>>>(ffnh, ffnh_bf,
                                                           ROWS * 4 * DM);
    gemm(ffnh_bf, w2, ffn_b2, ffno, ROWS, DM, 4 * DM, 4 * DM, DM, stream);
    add_ln_kernel<<<ROWS, 256, 0, stream>>>(ffno, out1, ffn_ln_g, ffn_ln_b,
                                            (float*)d_out);
}
